// AdaptiveAdjacencyMatrix_2379411882121
// MI455X (gfx1250) — compile-verified, matrix operand order checked by reading, not run on hardware
//
#include <hip/hip_runtime.h>

typedef __attribute__((ext_vector_type(16))) _Float16 v16h;
typedef __attribute__((ext_vector_type(8)))  float    v8f;
typedef __attribute__((ext_vector_type(4)))  float    f4;

#define M_DIM 1024
#define H_DIM 512

// ---------------------------------------------------------------------------
// Phase 1a: proj[b,m] = dot(h[b,m,:], w) via V_WMMA_F32_16X16X32_F16.
// 512 blocks x 4 waves = 2048 independent 16-row WMMA groups covering all
// 32*1024 rows -> enough memory-level parallelism to stream h (64 MB) at
// HBM rate. h is read exactly once: non-temporal loads.
// ---------------------------------------------------------------------------
__global__ __launch_bounds__(128) void aam_proj_kernel(
    const float* __restrict__ h, const float* __restrict__ w,
    float* __restrict__ proj)
{
    const int bb   = blockIdx.x >> 4;      // batch
    const int blk  = blockIdx.x & 15;      // sub-block within batch
    const int t    = threadIdx.x;
    const int lane = t & 31;
    const int wave = t >> 5;               // 0..3
    const int hs   = lane >> 4;            // K half-select (ISA A/B layout)
    const int mlo  = lane & 15;            // M within the 16-row tile

    const int m0 = (blk * 4 + wave) * 16;  // this wave's 16-row group
    const size_t rowbase = ((size_t)(bb * M_DIM + m0 + mlo)) * H_DIM;

    v8f c = {};
    for (int ck = 0; ck < 16; ++ck) {      // 16 chunks x K=32 covers H=512
        const int kb = ck * 32 + hs * 8;
        // A lane (row m0+mlo): halves 0..7 = K kb..kb+7, 8..15 = K kb+16..kb+23
        f4 a0 = __builtin_nontemporal_load((const f4*)(h + rowbase + kb));
        f4 a1 = __builtin_nontemporal_load((const f4*)(h + rowbase + kb) + 1);
        f4 a2 = __builtin_nontemporal_load((const f4*)(h + rowbase + kb + 16));
        f4 a3 = __builtin_nontemporal_load((const f4*)(h + rowbase + kb + 16) + 1);
        // B: same K pattern of w, identical for all 16 columns (broadcast);
        // w is 2 KB and hot in cache -> regular loads.
        const f4* bp = (const f4*)(w + kb);
        f4 b0 = bp[0], b1 = bp[1];
        const f4* bq = (const f4*)(w + kb + 16);
        f4 b2 = bq[0], b3 = bq[1];
        v16h av, bv;
#pragma unroll
        for (int q = 0; q < 4; ++q) {
            av[q]      = (_Float16)a0[q];
            av[4 + q]  = (_Float16)a1[q];
            av[8 + q]  = (_Float16)a2[q];
            av[12 + q] = (_Float16)a3[q];
            bv[q]      = (_Float16)b0[q];
            bv[4 + q]  = (_Float16)b1[q];
            bv[8 + q]  = (_Float16)b2[q];
            bv[12 + q] = (_Float16)b3[q];
        }
        c = __builtin_amdgcn_wmma_f32_16x16x32_f16(
                false, av, false, bv, (short)0, c, false, false);
    }
    // D columns are identical (B broadcast); lanes N==0 (lane 0 & 16) hold
    // rows m0 + hs*8 + p in c[p].
    if (mlo == 0) {
#pragma unroll
        for (int p = 0; p < 8; ++p)
            proj[bb * M_DIM + m0 + hs * 8 + p] = c[p];
    }
}

// ---------------------------------------------------------------------------
// Phase 1b: per-batch masked softmax vector (row-independent!):
//   p_j = exp(proj_j - mx)/Z ;  val[b,j] = [j<size] * p_j / (SV + 1e-10)
// where SV = sum_{j<size} p_j.  One block per batch, values held in registers.
// ---------------------------------------------------------------------------
__global__ __launch_bounds__(256) void aam_softmax_kernel(
    const float* __restrict__ proj, const int* __restrict__ sizes,
    float* __restrict__ val)
{
    __shared__ float s_red[256];

    const int bb = blockIdx.x;
    const int t  = threadIdx.x;
    const int size = sizes[bb];

    float pv[4];
#pragma unroll
    for (int k = 0; k < 4; ++k) pv[k] = proj[bb * M_DIM + t + 256 * k];

    // ---- max over 1024 ----
    float lmax = fmaxf(fmaxf(pv[0], pv[1]), fmaxf(pv[2], pv[3]));
    s_red[t] = lmax; __syncthreads();
    for (int s = 128; s > 0; s >>= 1) {
        if (t < s) s_red[t] = fmaxf(s_red[t], s_red[t + s]);
        __syncthreads();
    }
    const float mx = s_red[0];
    __syncthreads();

    // ---- exp, full sum Z, masked sum ----
    float e[4], lsum = 0.f, lmsk = 0.f;
#pragma unroll
    for (int k = 0; k < 4; ++k) {
        const int j = t + 256 * k;
        const float ev = __expf(pv[k] - mx);
        e[k] = ev; lsum += ev; if (j < size) lmsk += ev;
    }
    s_red[t] = lsum; __syncthreads();
    for (int s = 128; s > 0; s >>= 1) {
        if (t < s) s_red[t] += s_red[t + s];
        __syncthreads();
    }
    const float Z = s_red[0];
    __syncthreads();
    s_red[t] = lmsk; __syncthreads();
    for (int s = 128; s > 0; s >>= 1) {
        if (t < s) s_red[t] += s_red[t + s];
        __syncthreads();
    }
    const float SVe = s_red[0];

    // reference: p = e/Z; out = p / (sum_{valid} p + 1e-10)
    const float scale = 1.0f / (Z * (SVe / Z + 1e-10f));
#pragma unroll
    for (int k = 0; k < 4; ++k) {
        const int j = t + 256 * k;
        val[bb * M_DIM + j] = (j < size) ? e[k] * scale : 0.f;
    }
}

// ---------------------------------------------------------------------------
// Phase 2: one block per output row (b,i). Every valid row is the same 4 KB
// vector val[b,:] (pre-masked in columns); invalid rows are zero. 134 MB of
// 128-bit non-temporal streaming stores — the bandwidth floor (~5.8 us).
// ---------------------------------------------------------------------------
__global__ __launch_bounds__(256) void aam_broadcast_rows_kernel(
    const float* __restrict__ val, const int* __restrict__ sizes,
    float* __restrict__ out)
{
    const int idx = blockIdx.x;          // b*1024 + i
    const int bb  = idx >> 10;
    const int i   = idx & (M_DIM - 1);
    const int t   = threadIdx.x;

    f4 v = {0.f, 0.f, 0.f, 0.f};
    if (i < sizes[bb])
        v = ((const f4*)(val + bb * M_DIM))[t];   // 128 KB total, L2-resident

    f4* dst = (f4*)out + (((size_t)idx) << 8) + t;
    __builtin_nontemporal_store(v, dst);
}

extern "C" void kernel_launch(void* const* d_in, const int* in_sizes, int n_in,
                              void* d_out, int out_size, void* d_ws, size_t ws_size,
                              hipStream_t stream) {
    const float* h     = (const float*)d_in[0];
    const float* w     = (const float*)d_in[1];
    // d_in[2] = bias scalar: cancels inside the row softmax, unused.
    const int*   sizes = (const int*)d_in[3];
    float*       out   = (float*)d_out;

    float* proj = (float*)d_ws;                  // 32*1024 f32 = 128 KB
    float* val  = proj + 32 * M_DIM;             // 32*1024 f32 = 128 KB

    aam_proj_kernel<<<32 * 16, 128, 0, stream>>>(h, w, proj);
    aam_softmax_kernel<<<32, 256, 0, stream>>>(proj, sizes, val);
    aam_broadcast_rows_kernel<<<32 * M_DIM, 256, 0, stream>>>(val, sizes, out);
}
